// GCN_AE_variant1_37185826849263
// MI455X (gfx1250) — matrix operand order repaired and checked by reading, not compiled, not run on hardware
//
#include <hip/hip_runtime.h>
#include <hip/hip_bf16.h>

// ---------------------------------------------------------------------------
// GCN auto-encoder for MI455X (gfx1250, wave32).
//   - GCN dense GEMMs  -> v_wmma_f32_16x16x32_bf16 (fp32 accumulate)
//   - scatter-add      -> global_atomic_add_f32 (L2-resident, 192MB L2)
//   - 327MB We1 GEMV   -> coalesced streaming + global_prefetch_b8
// ---------------------------------------------------------------------------

#define N_NODES 10000
#define N_EDGES 320000
#define E_TOT   (N_EDGES + N_NODES)
#define F1 128
#define F2 64
#define F3 16
#define LATDIM 256
#define HVEC   (N_NODES * F3)   // 160000

typedef __attribute__((ext_vector_type(16))) __bf16 v16bf;
typedef __attribute__((ext_vector_type(8)))  float  v8f;

// ---------------------------------------------------------------- utilities
__global__ void k_zero(float* __restrict__ p, int n) {
  int i = blockIdx.x * blockDim.x + threadIdx.x;
  if (i < n) p[i] = 0.0f;
}

// ------------------------------------------------------------ degree / norm
__global__ void k_degree(const int* __restrict__ ei, float* __restrict__ deg) {
  int e = blockIdx.x * blockDim.x + threadIdx.x;
  if (e >= E_TOT) return;
  int d = (e < N_EDGES) ? ei[N_EDGES + e] : (e - N_EDGES);
  atomicAdd(&deg[d], 1.0f);
}

__global__ void k_dinv(float* __restrict__ deg) {
  int i = blockIdx.x * blockDim.x + threadIdx.x;
  if (i < N_NODES) deg[i] = rsqrtf(fmaxf(deg[i], 1.0f));
}

// --------------------------------------------------- layer1: outer product
__global__ void k_layer1(const float* __restrict__ x, const float* __restrict__ W1,
                         float* __restrict__ h) {
  int idx = blockIdx.x * blockDim.x + threadIdx.x;
  if (idx < N_NODES * F1) h[idx] = x[idx >> 7] * W1[idx & (F1 - 1)];
}

// ------------------------------------------- normalized scatter-add (A+I)h
template <int F>
__global__ void k_scatter(const int* __restrict__ ei, const float* __restrict__ dinv,
                          const float* __restrict__ h, float* __restrict__ out) {
  long long idx = (long long)blockIdx.x * blockDim.x + threadIdx.x;
  if (idx >= (long long)E_TOT * F) return;
  int e = (int)(idx >> __builtin_ctz(F));
  int f = (int)(idx & (F - 1));
  int s, d;
  if (e < N_EDGES) { s = ei[e]; d = ei[N_EDGES + e]; }
  else             { s = d = e - N_EDGES; }
  float v = h[(long long)s * F + f] * dinv[s] * dinv[d];
  atomicAdd(&out[(long long)d * F + f], v);
}

__global__ void k_bias_act(float* __restrict__ h, const float* __restrict__ b,
                           int total, int fmask, int relu) {
  int idx = blockIdx.x * blockDim.x + threadIdx.x;
  if (idx >= total) return;
  float v = h[idx] + b[idx & fmask];
  h[idx] = relu ? fmaxf(v, 0.0f) : v;
}

// ------------------------------------------------------- WMMA tile GEMM
// C[nrows,F] = A[nrows,K] @ B[K,F], row-major fp32 in/out, bf16 matrix core.
// One wave per 16x16 C tile; K stepped by 32 (v_wmma_f32_16x16x32_bf16).
template <int K, int F>
__global__ __launch_bounds__(32)
void k_gemm_wmma(const float* __restrict__ A, const float* __restrict__ B,
                 float* __restrict__ C) {
  const int ftiles = F / 16;
  int tile = blockIdx.x;
  int m0 = (tile / ftiles) * 16;
  int n0 = (tile % ftiles) * 16;
  int lane = threadIdx.x;        // 0..31, wave32
  int half = lane >> 4;          // lane group (0: lanes 0-15, 1: lanes 16-31)
  int l15  = lane & 15;

  v8f acc = {};
#pragma unroll
  for (int k0 = 0; k0 < K; k0 += 32) {
    // A operand: row m = l15; v16 elems 0..15 -> K = k0+16*half+[0..15]
    //            (same k order as B)
    v16bf a;
    const float* arow = A + (size_t)(m0 + l15) * K + k0 + half * 16;
#pragma unroll
    for (int j = 0; j < 8; ++j) {
      a[j]     = (__bf16)arow[j];
      a[j + 8] = (__bf16)arow[8 + j];
    }
    // B operand: col n = l15; lanes 0-15 hold K = k0+[0..15],
    //            lanes 16-31 hold K = k0+16+[0..15]
    v16bf b;
    const float* bcol = B + (size_t)(k0 + half * 16) * F + n0 + l15;
#pragma unroll
    for (int j = 0; j < 16; ++j) b[j] = (__bf16)bcol[(size_t)j * F];

    acc = __builtin_amdgcn_wmma_f32_16x16x32_bf16(
        /*neg_a=*/false, a, /*neg_b=*/false, b,
        /*c_mod=*/(short)0, acc, /*reuse_a=*/false, /*reuse_b=*/false);
  }
  // C/D layout: VGPR r -> row m0+r (lanes 0-15) / m0+r+8 (lanes 16-31)
#pragma unroll
  for (int r = 0; r < 8; ++r)
    C[(size_t)(m0 + r + half * 8) * F + n0 + l15] = acc[r];
}

// ----------------------------------- big streaming GEMV: acc[j] += x.We1[:,j]
// blockDim = 512 (one thread per output column), each block owns a row slab.
__global__ void k_gemv_acc(const float* __restrict__ x, const float* __restrict__ W,
                           float* __restrict__ acc, int rows_per_block, int total_rows) {
  const int ncols = 2 * LATDIM;  // 512
  int j  = threadIdx.x;
  int i0 = blockIdx.x * rows_per_block;
  int i1 = min(i0 + rows_per_block, total_rows);
  float s = 0.0f;
  for (int i = i0; i < i1; ++i) {
    __builtin_prefetch(&W[(size_t)(i + 8) * ncols + j], 0, 1);  // global_prefetch_b8
    s = fmaf(x[i], W[(size_t)i * ncols + j], s);
  }
  atomicAdd(&acc[j], s);
}

__global__ void k_vec_bias_relu(const float* __restrict__ in, const float* __restrict__ b,
                                float* __restrict__ out, int n) {
  int i = blockIdx.x * blockDim.x + threadIdx.x;
  if (i < n) out[i] = fmaxf(in[i] + b[i], 0.0f);
}

// -------------------------------------------- small MLP layer (x in LDS)
__global__ void k_mlp(const float* __restrict__ xin, const float* __restrict__ W,
                      const float* __restrict__ b, float* __restrict__ out,
                      int nin, int nout, int relu) {
  __shared__ float sx[512];
  for (int i = threadIdx.x; i < nin; i += blockDim.x) sx[i] = xin[i];
  __syncthreads();
  int j = blockIdx.x * blockDim.x + threadIdx.x;
  if (j >= nout) return;
  float s = b[j];
  for (int i = 0; i < nin; ++i) s = fmaf(sx[i], W[(size_t)i * nout + j], s);
  out[j] = relu ? fmaxf(s, 0.0f) : s;
}

// ---------------------------------------------------------------------------
extern "C" void kernel_launch(void* const* d_in, const int* in_sizes, int n_in,
                              void* d_out, int out_size, void* d_ws, size_t ws_size,
                              hipStream_t stream) {
  const float* x   = (const float*)d_in[0];
  const int*   ei  = (const int*)  d_in[1];   // [2,E] flat: src = ei[e], dst = ei[E+e]
  const float* W1  = (const float*)d_in[2];
  const float* b1  = (const float*)d_in[3];
  const float* W2  = (const float*)d_in[4];
  const float* b2  = (const float*)d_in[5];
  const float* W3  = (const float*)d_in[6];
  const float* b3  = (const float*)d_in[7];
  const float* We1 = (const float*)d_in[8];
  const float* be1 = (const float*)d_in[9];
  const float* We2 = (const float*)d_in[10];
  const float* be2 = (const float*)d_in[11];
  const float* Wd1 = (const float*)d_in[12];
  const float* bd1 = (const float*)d_in[13];
  const float* Wd2 = (const float*)d_in[14];
  const float* bd2 = (const float*)d_in[15];
  float* out = (float*)d_out;

  // workspace carve-up (~11 MB of fp32)
  float* ws = (float*)d_ws;
  size_t off = 0;
  float* dinv    = ws + off; off += N_NODES;
  float* bufA    = ws + off; off += (size_t)N_NODES * F1;
  float* bufB    = ws + off; off += (size_t)N_NODES * F1;
  float* bufC    = ws + off; off += (size_t)N_NODES * F3;
  float* enc_tmp = ws + off; off += 2 * LATDIM;
  float* z1      = ws + off; off += 2 * LATDIM;
  float* zv      = ws + off; off += LATDIM;
  float* h2      = ws + off; off += 2 * LATDIM;

  const int B256 = 256;
  auto blocks = [](long long n, int b) { return (unsigned)((n + b - 1) / b); };

  // degrees -> dinv
  k_zero<<<blocks(N_NODES, B256), B256, 0, stream>>>(dinv, N_NODES);
  k_degree<<<blocks(E_TOT, B256), B256, 0, stream>>>(ei, dinv);
  k_dinv<<<blocks(N_NODES, B256), B256, 0, stream>>>(dinv);

  // ---- GCN layer 1: h = x@W1 (outer product), aggregate, +b1, relu
  k_layer1<<<blocks((long long)N_NODES * F1, B256), B256, 0, stream>>>(x, W1, bufA);
  k_zero<<<blocks((long long)N_NODES * F1, B256), B256, 0, stream>>>(bufB, N_NODES * F1);
  k_scatter<F1><<<blocks((long long)E_TOT * F1, B256), B256, 0, stream>>>(ei, dinv, bufA, bufB);
  k_bias_act<<<blocks((long long)N_NODES * F1, B256), B256, 0, stream>>>(bufB, b1, N_NODES * F1, F1 - 1, 1);

  // ---- GCN layer 2: WMMA GEMM [N,128]x[128,64], aggregate, +b2, relu
  k_gemm_wmma<F1, F2><<<(N_NODES / 16) * (F2 / 16), 32, 0, stream>>>(bufB, W2, bufA);
  k_zero<<<blocks((long long)N_NODES * F2, B256), B256, 0, stream>>>(bufB, N_NODES * F2);
  k_scatter<F2><<<blocks((long long)E_TOT * F2, B256), B256, 0, stream>>>(ei, dinv, bufA, bufB);
  k_bias_act<<<blocks((long long)N_NODES * F2, B256), B256, 0, stream>>>(bufB, b2, N_NODES * F2, F2 - 1, 1);

  // ---- GCN layer 3: WMMA GEMM [N,64]x[64,16], aggregate, +b3 (no relu)
  k_gemm_wmma<F2, F3><<<(N_NODES / 16) * (F3 / 16), 32, 0, stream>>>(bufB, W3, bufA);
  k_zero<<<blocks((long long)N_NODES * F3, B256), B256, 0, stream>>>(bufC, N_NODES * F3);
  k_scatter<F3><<<blocks((long long)E_TOT * F3, B256), B256, 0, stream>>>(ei, dinv, bufA, bufC);
  k_bias_act<<<blocks((long long)N_NODES * F3, B256), B256, 0, stream>>>(bufC, b3, N_NODES * F3, F3 - 1, 0);

  // ---- encoder: z1 = relu(h@We1 + be1)   (327MB stream, BW-critical)
  k_zero<<<blocks(2 * LATDIM, B256), B256, 0, stream>>>(enc_tmp, 2 * LATDIM);
  k_gemv_acc<<<HVEC / 256, 2 * LATDIM, 0, stream>>>(bufC, We1, enc_tmp, 256, HVEC);
  k_vec_bias_relu<<<blocks(2 * LATDIM, B256), B256, 0, stream>>>(enc_tmp, be1, z1, 2 * LATDIM);
  // z = z1@We2 + be2
  k_mlp<<<1, B256, 0, stream>>>(z1, We2, be2, zv, 2 * LATDIM, LATDIM, 0);

  // ---- decoder
  k_mlp<<<2, B256, 0, stream>>>(zv, Wd1, bd1, h2, LATDIM, 2 * LATDIM, 1);
  k_mlp<<<blocks(N_NODES, B256), B256, 0, stream>>>(h2, Wd2, bd2, out, 2 * LATDIM, N_NODES, 0);
}
